// RecurrentDecoder_34943853920991
// MI455X (gfx1250) — matrix operand order repaired and checked by reading, not compiled, not run on hardware
//
#include <hip/hip_runtime.h>
#include <hip/hip_bf16.h>

typedef unsigned short u16;
typedef unsigned int   u32;
typedef unsigned long long u64;

typedef __attribute__((ext_vector_type(16))) __bf16 v16bf;
typedef __attribute__((ext_vector_type(8)))  float  v8f;
typedef __attribute__((ext_vector_type(4)))  u32    v4u;
typedef __attribute__((ext_vector_type(8)))  int    v8i;
typedef __attribute__((ext_vector_type(4)))  int    v4i;

#define B_  16
#define T_  255
#define E_  512
#define H_  1024
#define G_  3072     // 3*H
#define V_  32000
#define M_  (B_*T_)  // 4080

#define BM 128
#define BN 128
#define BK 32

// ---- workspace layout (bytes, 256-aligned) ----
#define OFF_TEACHER   0ull          // 4080*512 bf16        = 4,177,920
#define OFF_GKT       4177920ull    // 3072*512 bf16        = 3,145,728
#define OFF_GRKT      7323648ull    // 3072*1024 bf16       = 6,291,456
#define OFF_DWT       13615104ull   // 32000*1024 bf16      = 65,536,000
#define OFF_XP        79151104ull   // 4080*3072 f32        = 50,135,040
#define OFF_GRUOUT    129286144ull  // 4080*1024 bf16       = 8,355,840
#define OFF_HF32      137641984ull  // 16*1024 f32
#define OFF_HBF16     137707520ull  // 2*16*1024 bf16 (double buffer)
#define OFF_BAR       137773056ull  // barrier counters

// ---------------- helpers ----------------
__device__ inline u16 f2bf(float f) {
  __hip_bfloat16 h = __float2bfloat16(f);
  u16 u; __builtin_memcpy(&u, &h, 2); return u;
}

__device__ inline float sigmoidf_(float x) { return 1.0f / (1.0f + __expf(-x)); }

union FragCast { uint4 q[2]; v16bf v; };

// A-matrix 16x32 bf16 (ISA 7.12.2): lane row = lane%16;
// lane<16 holds K {0..7,16..23}, lane>=16 holds K {8..15,24..31}.
__device__ inline v16bf load_frag_a(const u16* base, int stride, int lane) {
  int r  = lane & 15;
  int kb = (lane < 16) ? 0 : 8;
  const u16* p = base + r * stride + kb;
  FragCast f;
  f.q[0] = *(const uint4*)(p);
  f.q[1] = *(const uint4*)(p + 16);
  return f.v;
}

// B-matrix 32x16 bf16, stored transposed BT[n][k]: lane col = lane%16;
// lane<16 holds K {0..7,16..23}, lane>=16 holds K {8..15,24..31}.
__device__ inline v16bf load_frag_b(const u16* base, int stride, int lane) {
  int n  = lane & 15;
  int kb = (lane < 16) ? 0 : 8;
  const u16* p = base + n * stride + kb;
  FragCast f;
  f.q[0] = *(const uint4*)(p);
  f.q[1] = *(const uint4*)(p + 16);
  return f.v;
}

__device__ inline v8f wmma_bf16(v16bf a, v16bf b, v8f c) {
  return __builtin_amdgcn_wmma_f32_16x16x32_bf16(false, a, false, b, (short)0, c,
                                                 false, false);
}

// Tensor Data Mover: 2D tile (BK x tile1) of bf16, D# per ISA ch.8.
// tensor_dim* given as remaining extent from the tile origin -> OOB rows read 0.
// This toolchain's builtin takes 6 args: (g0 v4u, g1 v8i, g2 v4i, g3 v4i, g4 v8i, cpol).
__device__ inline void tdm_load_2d(const u16* lds_dst, const u16* gsrc,
                                   u32 stride_elems, u32 dim1_rem, u32 dim0_rem,
                                   u32 tile1) {
  u32 lds_addr = (u32)(u64)lds_dst;           // low 32 bits = LDS byte offset
  u64 ga = (u64)gsrc;
  v4u g0;
  g0.x = 1u;                                  // count=1, user descriptor
  g0.y = lds_addr;
  g0.z = (u32)ga;
  g0.w = ((u32)(ga >> 32) & 0x01FFFFFFu) | 0x80000000u;  // addr[56:32] | type=2
  u32 d1 = (dim0_rem & 0xFFFFu) << 16;                    // tensor_dim0[15:0]
  u32 d2 = (dim0_rem >> 16) | ((dim1_rem & 0xFFFFu) << 16);
  u32 d3 = (dim1_rem >> 16) | ((u32)BK << 16);            // tile_dim0 = BK
  u32 d4 = tile1;                                         // tile_dim1, tile_dim2=0
  v8i g1 = { (int)0x00010000u /*data_size=2B*/, (int)d1, (int)d2, (int)d3,
             (int)d4, (int)stride_elems, 0, 0 };
  v4i g2 = { 0, 0, 0, 0 };
  v4i g3 = { 0, 0, 0, 0 };
  v8i g4 = { 0, 0, 0, 0, 0, 0, 0, 0 };
  __builtin_amdgcn_tensor_load_to_lds(g0, g1, g2, g3, g4, 0);
}

// ---------------- prep kernels ----------------
__global__ void k_prep_h(const float* __restrict__ latent, float* __restrict__ h_f32,
                         u16* __restrict__ h_bf16, u32* __restrict__ bar) {
  int i = blockIdx.x * blockDim.x + threadIdx.x;
  if (i < 2) bar[i * 64] = 0u;
  if (i < B_ * H_) {
    float v = latent[i];
    h_f32[i]  = v;
    h_bf16[i] = f2bf(v);
  }
}

__global__ void k_pack_teacher(const float* __restrict__ tout, u16* __restrict__ dst) {
  long long i = (long long)blockIdx.x * blockDim.x + threadIdx.x;
  if (i < (long long)M_ * E_) {
    int m = (int)(i >> 9), e = (int)(i & 511);
    int b = m / T_, t = m - b * T_;
    dst[i] = f2bf(tout[((long long)(b * 256 + t) << 9) + e]);
  }
}

__global__ void k_transpose_bf16(const float* __restrict__ src, u16* __restrict__ dst,
                                 int R, int Ccols) {
  long long i = (long long)blockIdx.x * blockDim.x + threadIdx.x;
  if (i < (long long)R * Ccols) {
    int r = (int)(i / Ccols);
    int c = (int)(i - (long long)r * Ccols);
    dst[(long long)c * R + r] = f2bf(src[i]);   // coalesced read, bf16 scatter
  }
}

// ---------------- main GEMM: C[M,N] = A[M,K] * BT[N,K]^T + bias ----------------
__global__ __launch_bounds__(256)
void k_gemm_bf16_tdm(const u16* __restrict__ A, const u16* __restrict__ BT,
                     const float* __restrict__ bias, float* __restrict__ C,
                     int M, int N, int K) {
  __shared__ u16 At[2][BM * BK];
  __shared__ u16 Bt[2][BN * BK];
  const int row0 = blockIdx.y * BM;
  const int col0 = blockIdx.x * BN;
  const int tid  = threadIdx.x;
  const int lane = tid & 31;
  const int wave = tid >> 5;
  const int wm   = wave >> 1;      // 0..3 -> 32 rows each
  const int wn   = wave & 1;       // 0..1 -> 64 cols each

  v8f acc[2][4];
  const v8f zf = {0.f,0.f,0.f,0.f,0.f,0.f,0.f,0.f};
#pragma unroll
  for (int i = 0; i < 2; ++i)
#pragma unroll
    for (int j = 0; j < 4; ++j) acc[i][j] = zf;

  const int niter = K / BK;
  if (wave == 0)
    tdm_load_2d(&At[0][0], A + (u64)row0 * K, (u32)K, (u32)(M - row0), (u32)K, BM);
  else if (wave == 1)
    tdm_load_2d(&Bt[0][0], BT + (u64)col0 * K, (u32)K, (u32)(N - col0), (u32)K, BN);

  for (int it = 0; it < niter; ++it) {
    if (wave < 2) __builtin_amdgcn_s_wait_tensorcnt(0);
    __syncthreads();                  // buffers for `it` ready; `it-1` consumers done
    const int cur = it & 1, nxt = cur ^ 1;
    if (it + 1 < niter) {             // prefetch next K-slice via TDM while computing
      const int k0 = (it + 1) * BK;
      if (wave == 0)
        tdm_load_2d(&At[nxt][0], A + (u64)row0 * K + k0, (u32)K,
                    (u32)(M - row0), (u32)(K - k0), BM);
      else if (wave == 1)
        tdm_load_2d(&Bt[nxt][0], BT + (u64)col0 * K + k0, (u32)K,
                    (u32)(N - col0), (u32)(K - k0), BN);
    }
    v16bf af[2], bf[4];
#pragma unroll
    for (int i = 0; i < 2; ++i)
      af[i] = load_frag_a(&At[cur][(wm * 32 + i * 16) * BK], BK, lane);
#pragma unroll
    for (int j = 0; j < 4; ++j)
      bf[j] = load_frag_b(&Bt[cur][(wn * 64 + j * 16) * BK], BK, lane);
#pragma unroll
    for (int i = 0; i < 2; ++i)
#pragma unroll
      for (int j = 0; j < 4; ++j)
        acc[i][j] = wmma_bf16(af[i], bf[j], acc[i][j]);
  }

  // C/D layout: M = v + (lane>=16 ? 8 : 0), N = lane%16
#pragma unroll
  for (int j = 0; j < 4; ++j) {
    const int c  = col0 + wn * 64 + j * 16 + (lane & 15);
    const float bv = bias[c];
#pragma unroll
    for (int i = 0; i < 2; ++i) {
      const int rbase = row0 + wm * 32 + i * 16 + ((lane >> 4) << 3);
#pragma unroll
      for (int v = 0; v < 8; ++v) {
        const int r = rbase + v;
        if (r < M) C[(u64)r * N + c] = acc[i][j][v] + bv;
      }
    }
  }
}

// ---------------- device-scope grid barrier ----------------
__device__ inline void grid_barrier(u32* bar, u32 nblk) {
  __threadfence();
  __syncthreads();
  if (threadIdx.x == 0) {
    u32* cnt = bar;
    u32* gen = bar + 64;
    u32 g = __hip_atomic_load(gen, __ATOMIC_ACQUIRE, __HIP_MEMORY_SCOPE_AGENT);
    u32 arrived = __hip_atomic_fetch_add(cnt, 1u, __ATOMIC_ACQ_REL, __HIP_MEMORY_SCOPE_AGENT);
    if (arrived == nblk - 1) {
      __hip_atomic_store(cnt, 0u, __ATOMIC_RELAXED, __HIP_MEMORY_SCOPE_AGENT);
      __hip_atomic_fetch_add(gen, 1u, __ATOMIC_RELEASE, __HIP_MEMORY_SCOPE_AGENT);
    } else {
      while (__hip_atomic_load(gen, __ATOMIC_ACQUIRE, __HIP_MEMORY_SCOPE_AGENT) == g)
        __builtin_amdgcn_s_sleep(1);
    }
  }
  __syncthreads();
}

// ---------------- persistent GRU scan: 8 WGs, 255 steps ----------------
__global__ __launch_bounds__(256)
void k_gru_scan(const u16* __restrict__ grkT, const float* __restrict__ xp,
                const float* __restrict__ rbias, float* __restrict__ h_f32,
                u16* __restrict__ h_bf16, u16* __restrict__ gru_out,
                u32* __restrict__ bar) {
  const int lane = threadIdx.x & 31;
  const int wave = threadIdx.x >> 5;
  const int cw   = blockIdx.x * 128 + wave * 16;   // 16 h-columns per wave
  const int j    = cw + (lane & 15);
  const int mrow8 = (lane >> 4) << 3;
  const float rbz = rbias[j];
  const float rbr = rbias[H_ + j];
  const float rbh = rbias[2 * H_ + j];
  const v8f zf = {0.f,0.f,0.f,0.f,0.f,0.f,0.f,0.f};

  for (int t = 0; t < T_; ++t) {
    const u16* hb = h_bf16 + (u64)(t & 1) * (B_ * H_);
    v8f az = zf, ar = zf, ah = zf;
    // rec = h(16xH) @ Wr(Hx3H): one 16x16 tile per gate for this wave
    for (int k = 0; k < H_; k += BK) {
      v16bf a  = load_frag_a(hb + k, H_, lane);
      v16bf bz = load_frag_b(grkT + (u64)(cw) * H_ + k,            H_, lane);
      v16bf br = load_frag_b(grkT + (u64)(H_ + cw) * H_ + k,       H_, lane);
      v16bf bh = load_frag_b(grkT + (u64)(2 * H_ + cw) * H_ + k,   H_, lane);
      az = wmma_bf16(a, bz, az);
      ar = wmma_bf16(a, br, ar);
      ah = wmma_bf16(a, bh, ah);
    }
    u16* hbn = h_bf16 + (u64)((t + 1) & 1) * (B_ * H_);
#pragma unroll
    for (int v = 0; v < 8; ++v) {
      const int b = v + mrow8;
      const u64 m = (u64)b * T_ + t;
      const float xz = xp[m * G_ + j];
      const float xr = xp[m * G_ + H_ + j];
      const float xh = xp[m * G_ + 2 * H_ + j];
      const float z  = sigmoidf_(xz + az[v] + rbz);
      const float r  = sigmoidf_(xr + ar[v] + rbr);
      const float hh = tanhf(xh + r * (ah[v] + rbh));
      const int   hi = b * H_ + j;
      const float hp = h_f32[hi];
      const float hn = z * hp + (1.0f - z) * hh;
      h_f32[hi] = hn;                  // private f32 carry (owner-only columns)
      const u16 hu = f2bf(hn);
      hbn[hi] = hu;                    // shared bf16 state for next step's GEMM
      gru_out[m * H_ + j] = hu;        // logits-GEMM A operand
    }
    grid_barrier(bar, gridDim.x);
  }
}

// ---------------- launcher ----------------
extern "C" void kernel_launch(void* const* d_in, const int* in_sizes, int n_in,
                              void* d_out, int out_size, void* d_ws, size_t ws_size,
                              hipStream_t stream) {
  const float* latent = (const float*)d_in[0];
  const float* tout   = (const float*)d_in[1];
  const float* gk     = (const float*)d_in[2];
  const float* grk    = (const float*)d_in[3];
  const float* gbias  = (const float*)d_in[4];   // [2][3072]
  const float* dw     = (const float*)d_in[5];
  const float* db     = (const float*)d_in[6];
  float* out = (float*)d_out;
  char*  ws  = (char*)d_ws;

  u16*   teacherB = (u16*)(ws + OFF_TEACHER);
  u16*   gkT      = (u16*)(ws + OFF_GKT);
  u16*   grkT     = (u16*)(ws + OFF_GRKT);
  u16*   dwT      = (u16*)(ws + OFF_DWT);
  float* xprj     = (float*)(ws + OFF_XP);
  u16*   gruout   = (u16*)(ws + OFF_GRUOUT);
  float* h_f32    = (float*)(ws + OFF_HF32);
  u16*   h_bf16   = (u16*)(ws + OFF_HBF16);
  u32*   bar      = (u32*)(ws + OFF_BAR);

  // 1) prep / precision conversion / weight transposes
  k_prep_h<<<(B_ * H_ + 255) / 256, 256, 0, stream>>>(latent, h_f32, h_bf16, bar);
  k_pack_teacher<<<(M_ * E_ + 255) / 256, 256, 0, stream>>>(tout, teacherB);
  k_transpose_bf16<<<(E_ * G_ + 255) / 256, 256, 0, stream>>>(gk, gkT, E_, G_);
  k_transpose_bf16<<<(H_ * G_ + 255) / 256, 256, 0, stream>>>(grk, grkT, H_, G_);
  k_transpose_bf16<<<((long long)H_ * V_ + 255) / 256, 256, 0, stream>>>(dw, dwT, H_, V_);

  // 2) x_proj = teacher @ Wx + b0   (M=4080, N=3072, K=512)
  dim3 g1(G_ / BN, (M_ + BM - 1) / BM);
  k_gemm_bf16_tdm<<<g1, 256, 0, stream>>>(teacherB, gkT, gbias, xprj, M_, G_, E_);

  // 3) sequential GRU scan (persistent, grid-barrier per step)
  k_gru_scan<<<8, 256, 0, stream>>>(grkT, xprj, gbias + G_, h_f32, h_bf16, gruout, bar);

  // 4) logits = gru_out @ Wd + bd   (M=4080, N=32000, K=1024)
  dim3 g3(V_ / BN, (M_ + BM - 1) / BM);
  k_gemm_bf16_tdm<<<g3, 256, 0, stream>>>(gruout, dwT, db, out, M_, V_, H_);

  (void)in_sizes; (void)n_in; (void)out_size; (void)ws_size;
}
